// ContConv1d_11673721111189
// MI455X (gfx1250) — matrix operand order repaired and checked by reading, not compiled, not run on hardware
//
#include <hip/hip_runtime.h>

typedef __attribute__((ext_vector_type(16))) __bf16 v16bf;
typedef __attribute__((ext_vector_type(8)))  float  v8f;

#define BS    4
#define LSEQ  2048
#define CIN   32
#define COUT  32
#define HID   64
#define KW    8
#define PADL  7
#define TILES ((BS * LSEQ) / 16)   // 512 row-tiles of 16 timesteps
#define WAVES 4
#define NBLK  (TILES / WAVES)      // 128 workgroups
// LDS (bf16 units): W2 ops 65536 | W1 ops 2048 | b2 ops 1024 | Wskip ops 1024 | H 4*8*1024
#define SMEM_BF16 (65536 + 2048 + 1024 + 1024 + 32768)
#define SMEM_BYTES (SMEM_BF16 * 2 + WAVES * 64 * 4)

// A-operand (16-bit 16x32) K index for lane l, half-slot s (ISA 7.12.2)
__device__ __forceinline__ int kmapA(int l, int s) {
  return ((s < 8) ? s : (s + 8)) + ((l & 16) ? 8 : 0);
}
// B-operand (16-bit 32x16) K index: same order as A (lanes 0-15 K=0..7,16..23; lanes 16-31 K=8..15,24..31)
__device__ __forceinline__ int kmapB(int l, int s) {
  return ((s < 8) ? s : (s + 8)) + ((l & 16) ? 8 : 0);
}

__device__ __forceinline__ v16bf splat_bf(__bf16 x) {
  v16bf v;
#pragma unroll
  for (int s = 0; s < 16; ++s) v[s] = x;
  return v;
}

__global__ __launch_bounds__(128) void cc1d_main(
    const float* __restrict__ times, const float* __restrict__ feat,
    const unsigned char* __restrict__ npm,
    const float* __restrict__ W1, const float* __restrict__ b1,
    const float* __restrict__ W2, const float* __restrict__ b2,
    const float* __restrict__ Wsk, const float* __restrict__ bsk,
    float* __restrict__ out, float* __restrict__ partials)
{
  extern __shared__ __align__(32) char smem_raw[];
  __bf16* sm_w2 = (__bf16*)smem_raw;        // 64 j * 2 halves * 512 entries
  __bf16* sm_w1 = sm_w2 + 65536;            // 4 nt blocks * 512
  __bf16* sm_b2 = sm_w1 + 2048;             // 2 blocks * 512
  __bf16* sm_ws = sm_b2 + 1024;             // 2 blocks * 512
  __bf16* sm_H  = sm_ws + 1024;             // per-wave: 8 tiles of 16x64 bf16
  float*  sm_red = (float*)(sm_H + 32768);  // 4 waves * 32ch * {sum,sumsq}

  const int tid = threadIdx.x;

  // ---- cooperative swizzle of weights into WMMA-B operand order ----
  for (int e = tid; e < 65536; e += 128) {            // W2[j, c*32+o] -> B_{j,half}
    int ob = e >> 9, w = e & 511;
    int l = w >> 4, s = w & 15;
    int j = ob >> 1, half = ob & 1;
    int kk = kmapB(l, s);
    int n  = (l & 15) + half * 16;
    sm_w2[e] = (__bf16)W2[j * (CIN * COUT) + kk * COUT + n];
  }
  for (int e = tid; e < 2048; e += 128) {             // W1[i, j]
    int ob = e >> 9, w = e & 511;
    int l = w >> 4, s = w & 15;
    int kk = kmapB(l, s);
    sm_w1[e] = (__bf16)W1[kk * HID + ob * 16 + (l & 15)];
  }
  for (int e = tid; e < 1024; e += 128) {             // b2 as (32x32) matrix
    int ob = e >> 9, w = e & 511;
    int l = w >> 4, s = w & 15;
    int kk = kmapB(l, s);
    sm_b2[e] = (__bf16)b2[kk * COUT + ob * 16 + (l & 15)];
  }
  for (int e = tid; e < 1024; e += 128) {             // W_skip (32x32)
    int ob = e >> 9, w = e & 511;
    int l = w >> 4, s = w & 15;
    int kk = kmapB(l, s);
    sm_ws[e] = (__bf16)Wsk[kk * COUT + ob * 16 + (l & 15)];
  }
  __syncthreads();

  const int wv = tid >> 5;
  const int l  = tid & 31;
  const int tileId = blockIdx.x * WAVES + wv;
  const int b  = tileId >> 7;                  // 128 tiles per batch
  const int t0 = (tileId & 127) << 4;
  const int p  = l & 15;                       // matrix row owned by this lane
  const int t  = t0 + p;
  const int mbase = (l & 16) ? 8 : 0;          // C/D row base for this lane group
  const int cbase = (l & 16) ? 8 : 0;          // A-operand channel base

  // preload W1 operand blocks + biases
  v16bf Bw1[4];
#pragma unroll
  for (int nt = 0; nt < 4; ++nt)
    Bw1[nt] = *(const v16bf*)(sm_w1 + nt * 512 + l * 16);
  float b1g[4];
#pragma unroll
  for (int nt = 0; nt < 4; ++nt) b1g[nt] = b1[nt * 16 + (l & 15)];

  // per-lane temporal-encoding constants (depend only on lane/slot)
  float invp[16]; int odd[16];
#pragma unroll
  for (int s = 0; s < 16; ++s) {
    int i = kmapA(l, s);
    invp[s] = __expf(-0.5756462732485114f * (float)(i >> 1)); // 10000^(-(i/2)/16)
    odd[s]  = i & 1;
  }

  const float tcur = times[b * LSEQ + t];
  const int maskT = npm[b * LSEQ + t] ? 1 : 0;

  // ---------------- Phase A: h tiles (all k) + feature fragments ----------------
  v16bf fkb[KW];                 // masked gathered features, bf16, A-slot order
  float fsum[16];
#pragma unroll
  for (int s = 0; s < 16; ++s) fsum[s] = 0.f;

  for (int k = 0; k < KW; ++k) {
    const int tg    = t - PADL + k;
    const int valid = tg >= 0;
    const int tgc   = valid ? tg : 0;
    const int m     = (valid && npm[b * LSEQ + tgc] && maskT) ? 1 : 0;
    const float dt  = m ? (tcur - times[b * LSEQ + tgc]) : 0.0f;

    // temporal encoding tile -> A operand (16x32 bf16)
    v16bf aTE;
#pragma unroll
    for (int s = 0; s < 16; ++s) {
      float r = dt * invp[s];
      float v = odd[s] ? __cosf(r) : __sinf(r);
      aTE[s] = (__bf16)v;
    }

    // h = relu(te @ W1 + b1): 4 WMMAs, write H_k tile (16x64 bf16) to LDS
#pragma unroll
    for (int nt = 0; nt < 4; ++nt) {
      v8f c = {0.f,0.f,0.f,0.f,0.f,0.f,0.f,0.f};
      c = __builtin_amdgcn_wmma_f32_16x16x32_bf16(false, aTE, false, Bw1[nt],
                                                  (short)0, c, false, false);
      const int jcol = nt * 16 + (l & 15);
      const float bj = b1g[nt];
#pragma unroll
      for (int r8 = 0; r8 < 8; ++r8) {
        float hv = c[r8] + bj;
        hv = hv > 0.f ? hv : 0.f;
        sm_H[wv * 8192 + k * 1024 + (mbase + r8) * HID + jcol] = (__bf16)hv;
      }
    }

    // masked gathered features -> packed bf16 fragment
    {
      const float* fg = feat + (size_t)(b * LSEQ + tgc) * CIN + cbase;
      const float mf = (float)m;
#pragma unroll
      for (int s = 0; s < 8; ++s) {
        float v0 = fg[s]      * mf;
        float v1 = fg[16 + s] * mf;
        fkb[k][s]     = (__bf16)v0;
        fkb[k][8 + s] = (__bf16)v1;
        fsum[s]     += v0;
        fsum[8 + s] += v1;
      }
    }
  }

  // four accumulators: even/odd j columns to break WMMA dependency chains
  v8f accE0 = {0.f,0.f,0.f,0.f,0.f,0.f,0.f,0.f};
  v8f accE1 = {0.f,0.f,0.f,0.f,0.f,0.f,0.f,0.f};
  v8f accO0 = {0.f,0.f,0.f,0.f,0.f,0.f,0.f,0.f};
  v8f accO1 = {0.f,0.f,0.f,0.f,0.f,0.f,0.f,0.f};

  // b2 term: (sum_k masked feat) @ b2-matrix
  {
    v16bf a2;
#pragma unroll
    for (int s = 0; s < 16; ++s) a2[s] = (__bf16)fsum[s];
    v16bf bb0 = *(const v16bf*)(sm_b2 + 0   + l * 16);
    v16bf bb1 = *(const v16bf*)(sm_b2 + 512 + l * 16);
    accE0 = __builtin_amdgcn_wmma_f32_16x16x32_bf16(false, a2, false, bb0,
                                                    (short)0, accE0, false, false);
    accE1 = __builtin_amdgcn_wmma_f32_16x16x32_bf16(false, a2, false, bb1,
                                                    (short)0, accE1, false, false);
  }
  // skip path: raw features @ W_skip
  {
    const float* fr = feat + (size_t)(b * LSEQ + t) * CIN + cbase;
    v16bf a2;
#pragma unroll
    for (int s = 0; s < 8; ++s) {
      a2[s]     = (__bf16)fr[s];
      a2[8 + s] = (__bf16)fr[16 + s];
    }
    v16bf bb0 = *(const v16bf*)(sm_ws + 0   + l * 16);
    v16bf bb1 = *(const v16bf*)(sm_ws + 512 + l * 16);
    accO0 = __builtin_amdgcn_wmma_f32_16x16x32_bf16(false, a2, false, bb0,
                                                    (short)0, accO0, false, false);
    accO1 = __builtin_amdgcn_wmma_f32_16x16x32_bf16(false, a2, false, bb1,
                                                    (short)0, accO1, false, false);
  }

  // ---------------- Phase B: j-outer U-GEMM, B operands loaded once ----------------
  const __bf16* Hbase = sm_H + wv * 8192 + p * HID;   // + k*1024 + j
  for (int j = 0; j < HID; j += 2) {
    v16bf bE0 = *(const v16bf*)(sm_w2 + ((j + 0) * 2 + 0) * 512 + l * 16);
    v16bf bE1 = *(const v16bf*)(sm_w2 + ((j + 0) * 2 + 1) * 512 + l * 16);
    v16bf bO0 = *(const v16bf*)(sm_w2 + ((j + 1) * 2 + 0) * 512 + l * 16);
    v16bf bO1 = *(const v16bf*)(sm_w2 + ((j + 1) * 2 + 1) * 512 + l * 16);
#pragma unroll
    for (int k = 0; k < KW; ++k) {
      const unsigned int w = *(const unsigned int*)(Hbase + k * 1024 + j);
      const __bf16 h0 = __builtin_bit_cast(__bf16, (unsigned short)(w & 0xffffu));
      const __bf16 h1 = __builtin_bit_cast(__bf16, (unsigned short)(w >> 16));
      v16bf a0 = fkb[k] * splat_bf(h0);         // -> v_pk_mul_bf16
      accE0 = __builtin_amdgcn_wmma_f32_16x16x32_bf16(false, a0, false, bE0,
                                                      (short)0, accE0, false, false);
      accE1 = __builtin_amdgcn_wmma_f32_16x16x32_bf16(false, a0, false, bE1,
                                                      (short)0, accE1, false, false);
      v16bf a1 = fkb[k] * splat_bf(h1);
      accO0 = __builtin_amdgcn_wmma_f32_16x16x32_bf16(false, a1, false, bO0,
                                                      (short)0, accO0, false, false);
      accO1 = __builtin_amdgcn_wmma_f32_16x16x32_bf16(false, a1, false, bO1,
                                                      (short)0, accO1, false, false);
    }
  }

  const v8f acc0 = accE0 + accO0;
  const v8f acc1 = accE1 + accO1;

  // + b_skip, store pre-norm output, accumulate per-channel sum/sumsq
  const float bs0 = bsk[l & 15];
  const float bs1 = bsk[16 + (l & 15)];
  float s0 = 0.f, q0 = 0.f, s1 = 0.f, q1 = 0.f;
#pragma unroll
  for (int r8 = 0; r8 < 8; ++r8) {
    const int row = t0 + mbase + r8;
    const float v0 = acc0[r8] + bs0;
    const float v1 = acc1[r8] + bs1;
    out[(size_t)(b * LSEQ + row) * COUT + (l & 15)] = v0;
    out[(size_t)(b * LSEQ + row) * COUT + 16 + (l & 15)] = v1;
    s0 += v0; q0 += v0 * v0;
    s1 += v1; q1 += v1 * v1;
  }
  s0 += __shfl_xor(s0, 16, 32);
  q0 += __shfl_xor(q0, 16, 32);
  s1 += __shfl_xor(s1, 16, 32);
  q1 += __shfl_xor(q1, 16, 32);
  if (l < 16) {
    sm_red[wv * 64 + l * 2 + 0] = s0;
    sm_red[wv * 64 + l * 2 + 1] = q0;
    sm_red[wv * 64 + (16 + l) * 2 + 0] = s1;
    sm_red[wv * 64 + (16 + l) * 2 + 1] = q1;
  }
  __syncthreads();
  if (tid < 32) {                // deterministic fixed-order block reduction
    float S = 0.f, Q = 0.f;
#pragma unroll
    for (int w2 = 0; w2 < WAVES; ++w2) {
      S += sm_red[w2 * 64 + tid * 2 + 0];
      Q += sm_red[w2 * 64 + tid * 2 + 1];
    }
    partials[blockIdx.x * 64 + tid * 2 + 0] = S;
    partials[blockIdx.x * 64 + tid * 2 + 1] = Q;
  }
}

__global__ void cc1d_stats(const float* __restrict__ partials, float* __restrict__ stats) {
  const int ch = threadIdx.x;  // 32 threads
  float S = 0.f, Q = 0.f;
  for (int blk = 0; blk < NBLK; ++blk) {
    S += partials[blk * 64 + ch * 2 + 0];
    Q += partials[blk * 64 + ch * 2 + 1];
  }
  const float n = (float)(BS * LSEQ);
  const float mean = S / n;
  const float var  = Q / n - mean * mean;
  stats[ch] = mean;
  stats[32 + ch] = rsqrtf(var + 1e-5f);
}

__global__ void cc1d_norm(float* __restrict__ out, const float* __restrict__ stats,
                          const float* __restrict__ gamma, const float* __restrict__ beta) {
  const int i = blockIdx.x * blockDim.x + threadIdx.x;
  if (i < BS * LSEQ * COUT) {
    const int o = i & 31;
    const float v = out[i];
    out[i] = gamma[o] * (v - stats[o]) * stats[32 + o] + beta[o];
  }
}

extern "C" void kernel_launch(void* const* d_in, const int* in_sizes, int n_in,
                              void* d_out, int out_size, void* d_ws, size_t ws_size,
                              hipStream_t stream) {
  const float* times          = (const float*)d_in[0];
  const float* features       = (const float*)d_in[1];
  const unsigned char* npm    = (const unsigned char*)d_in[2];
  const float* W1             = (const float*)d_in[3];
  const float* b1             = (const float*)d_in[4];
  const float* W2             = (const float*)d_in[5];
  const float* b2             = (const float*)d_in[6];
  const float* Wsk            = (const float*)d_in[7];
  const float* bsk            = (const float*)d_in[8];
  const float* gamma          = (const float*)d_in[9];
  const float* beta           = (const float*)d_in[10];
  float* out = (float*)d_out;
  float* partials = (float*)d_ws;               // NBLK*64 floats
  float* stats    = partials + NBLK * 64;       // 64 floats

  cc1d_main<<<NBLK, 128, SMEM_BYTES, stream>>>(times, features, npm, W1, b1, W2,
                                               b2, Wsk, bsk, out, partials);
  cc1d_stats<<<1, 32, 0, stream>>>(partials, stats);
  cc1d_norm<<<(BS * LSEQ * COUT + 255) / 256, 256, 0, stream>>>(out, stats, gamma, beta);
}
